// SpatialAttention_40037685133357
// MI455X (gfx1250) — compile-verified, matrix operand order checked by reading, not run on hardware
//
#include <hip/hip_runtime.h>
#include <hip/hip_bf16.h>

typedef __attribute__((ext_vector_type(16))) _Float16 v16h;
typedef __attribute__((ext_vector_type(8)))  float    v8f;

#define HIDDEN 512
#define HEADS  8
#define NNODES 2048
#define BATCH  2
#define HD     64
#define ROWS   (BATCH * NNODES)      // 4096
#define MASKW  (NNODES / 32)         // 64 words per row
#define KT     (HIDDEN / 32)         // 16 k-tiles
#define NT     (HIDDEN / 16)         // 32 n-tiles
#define RT     (ROWS / 16)           // 256 row-tiles

// K-index for half-element i (0..15) of a 16-bit 16x32 A/B WMMA fragment,
// per CDNA5 ISA 7.12.2: VGPR p holds K pairs; lane-half selects +8 within group.
__device__ __forceinline__ int wmma_k16(int i, int half) {
    int p = i >> 1;
    return ((p & 4) << 2) + ((p & 3) << 1) + (half << 3) + (i & 1);
}

__device__ __forceinline__ float red_max16(float v) {
    v = fmaxf(v, __shfl_xor(v, 1, 32));
    v = fmaxf(v, __shfl_xor(v, 2, 32));
    v = fmaxf(v, __shfl_xor(v, 4, 32));
    v = fmaxf(v, __shfl_xor(v, 8, 32));
    return v;
}
__device__ __forceinline__ float red_sum16(float v) {
    v += __shfl_xor(v, 1, 32);
    v += __shfl_xor(v, 2, 32);
    v += __shfl_xor(v, 4, 32);
    v += __shfl_xor(v, 8, 32);
    return v;
}

// ---------------- mask build ----------------
__global__ void build_mask_kernel(const long long* __restrict__ ei, int nedges,
                                  unsigned* __restrict__ mask) {
    int idx = blockIdx.x * blockDim.x + threadIdx.x;
    if (idx >= nedges) return;
    int r = (int)ei[idx];            // query row
    int c = (int)ei[nedges + idx];   // key col
    atomicOr(&mask[r * MASKW + (c >> 5)], 1u << (c & 31));
}

// ---------------- repack: x (f32 row-major) -> f16 A-fragment order ----------------
// layout: [rowtile RT][ktile KT][lane 32][16 halves]
__global__ void pack_x_kernel(const float* __restrict__ X, _Float16* __restrict__ Xp) {
    int idx = blockIdx.x * blockDim.x + threadIdx.x;     // < RT*KT*32*16 = 2M
    int i    = idx & 15;
    int lane = (idx >> 4) & 31;
    int kt   = (idx >> 9) & (KT - 1);
    int rt   = idx >> 13;
    Xp[idx] = (_Float16)X[(size_t)(rt * 16 + (lane & 15)) * HIDDEN +
                          kt * 32 + wmma_k16(i, lane >> 4)];
}

// ---------------- repack: W (f32 [512k][512n]) -> f16 B-fragment order ----------------
// layout: [ktile KT][ntile NT][lane 32][16 halves]
__global__ void pack_w_kernel(const float* __restrict__ W, _Float16* __restrict__ Wp) {
    int idx = blockIdx.x * blockDim.x + threadIdx.x;     // < KT*NT*32*16 = 256K
    int i    = idx & 15;
    int lane = (idx >> 4) & 31;
    int nt   = (idx >> 9) & (NT - 1);
    int kt   = idx >> 14;
    Wp[idx] = (_Float16)W[(size_t)(kt * 32 + wmma_k16(i, lane >> 4)) * HIDDEN +
                          nt * 16 + (lane & 15)];
}

// ---------------- repack: K [bh][n][64] -> B-fragment order for Q.K^T ----------------
// layout: [bh 16][mtile 128][kc 2][lane 32][16 halves]
__global__ void pack_kb_kernel(const _Float16* __restrict__ K, _Float16* __restrict__ Kp) {
    int idx = blockIdx.x * blockDim.x + threadIdx.x;     // < 16*128*2*32*16 = 2M
    int i    = idx & 15;
    int lane = (idx >> 4) & 31;
    int kc   = (idx >> 9) & 1;
    int mt   = (idx >> 10) & 127;
    int bh   = idx >> 17;
    Kp[idx] = K[((size_t)bh * NNODES + mt * 16 + (lane & 15)) * HD +
                kc * 32 + wmma_k16(i, lane >> 4)];
}

// ---------------- repack: V [bh][n][64] -> B-fragment order for P.V ----------------
// layout: [bh 16][mstep 64][db 4][lane 32][16 halves]
__global__ void pack_vb_kernel(const _Float16* __restrict__ V, _Float16* __restrict__ Vp) {
    int idx = blockIdx.x * blockDim.x + threadIdx.x;     // < 16*64*4*32*16 = 2M
    int i    = idx & 15;
    int lane = (idx >> 4) & 31;
    int db   = (idx >> 9) & 3;
    int ms   = (idx >> 11) & 63;
    int bh   = idx >> 17;
    Vp[idx] = V[((size_t)bh * NNODES + ms * 32 + wmma_k16(i, lane >> 4)) * HD +
                db * 16 + (lane & 15)];
}

// ---------------- QKV projection: packed-f16 WMMA GEMM ----------------
// grid.x = RT; block = 256 (8 waves); wave w = head w (64 cols).
__global__ void gemm_proj_kernel(const _Float16* __restrict__ Xp, const _Float16* __restrict__ Wp,
                                 const float* __restrict__ bias, _Float16* __restrict__ out) {
    const int lane = threadIdx.x & 31;
    const int wave = threadIdx.x >> 5;
    const int half = lane >> 4;
    const int l15  = lane & 15;
    const int row0 = blockIdx.x * 16;
    const int col0 = wave * 64;

    v8f acc[4] = {};
#pragma unroll 1
    for (int kt = 0; kt < KT; ++kt) {
        const v16h a = *(const v16h*)(Xp + (((size_t)blockIdx.x * KT + kt) * 32 + lane) * 16);
#pragma unroll
        for (int db = 0; db < 4; ++db) {
            const int nt = (col0 >> 4) + db;
            const v16h b = *(const v16h*)(Wp + (((size_t)kt * NT + nt) * 32 + lane) * 16);
            acc[db] = __builtin_amdgcn_wmma_f32_16x16x32_f16(
                false, a, false, b, (short)0, acc[db], false, false);
        }
    }
#pragma unroll
    for (int db = 0; db < 4; ++db) {
        const int col = col0 + db * 16 + l15;
        const float bv = bias[col];
        const int h = col >> 6, d = col & 63;
#pragma unroll
        for (int r = 0; r < 8; ++r) {
            const int rg = row0 + r + 8 * half;     // global row in [0,4096)
            const int b  = rg >> 11, n = rg & (NNODES - 1);
            out[((size_t)((b << 3) + h) * NNODES + n) * HD + d] =
                (_Float16)(acc[db][r] + bv);
        }
    }
}

// ---------------- flash attention: 1 wave per (b,h,16-row tile) ----------------
__global__ void attention_kernel(const _Float16* __restrict__ Q, const _Float16* __restrict__ Kp,
                                 const _Float16* __restrict__ Vp, const unsigned* __restrict__ mask,
                                 _Float16* __restrict__ AttP) {
    __shared__ _Float16 pbuf[16 * 64];
    const int lane = threadIdx.x;
    const int half = lane >> 4;
    const int l15  = lane & 15;
    const int n0 = blockIdx.x * 16;
    const int bh = blockIdx.z * HEADS + blockIdx.y;
    const float scale = 0.125f;      // 1/sqrt(64)

    // Q tile as two A-fragments (K = 0..31, 32..63)
    v16h qa[2];
    {
        const _Float16* qr = Q + ((size_t)bh * NNODES + n0 + l15) * HD;
#pragma unroll
        for (int kc = 0; kc < 2; ++kc)
#pragma unroll
            for (int i = 0; i < 8; ++i) {
                int k = kc * 32 + wmma_k16(2 * i, half);
                qa[kc][2 * i]     = qr[k];
                qa[kc][2 * i + 1] = qr[k + 1];
            }
    }

    const _Float16* kbase = Kp + (size_t)bh * 128 * 2 * 32 * 16;
    const _Float16* vbase = Vp + (size_t)bh * 64 * 4 * 32 * 16;

    v8f acc[4] = {};                 // attended 16 x 64 accumulator
    float mst[8], lst[8];
#pragma unroll
    for (int r = 0; r < 8; ++r) { mst[r] = -1e30f; lst[r] = 0.f; }

#pragma unroll 1
    for (int m0 = 0; m0 < NNODES; m0 += 32) {
        const int mt = m0 >> 4;      // two 16-col tiles: mt, mt+1
        const int ms = m0 >> 5;

        // prefetch next iteration's K/V fragments into cache (global_prefetch_b8)
        if (m0 + 32 < NNODES) {
            __builtin_prefetch(kbase + (((size_t)(mt + 2) * 2) * 32 + lane) * 16, 0, 0);
            __builtin_prefetch(vbase + (((size_t)(ms + 1) * 4) * 32 + lane) * 16, 0, 0);
        }

        // scores: two 16x16 tiles from packed-B K fragments
        v8f s[2];
#pragma unroll
        for (int mh = 0; mh < 2; ++mh) {
            const v16h b0 = *(const v16h*)(kbase + (((size_t)(mt + mh) * 2 + 0) * 32 + lane) * 16);
            const v16h b1 = *(const v16h*)(kbase + (((size_t)(mt + mh) * 2 + 1) * 32 + lane) * 16);
            v8f z = {};
            z = __builtin_amdgcn_wmma_f32_16x16x32_f16(false, qa[0], false, b0, (short)0, z, false, false);
            z = __builtin_amdgcn_wmma_f32_16x16x32_f16(false, qa[1], false, b1, (short)0, z, false, false);
            s[mh] = z;
        }

        // mask word for this row / 32-col window (lane l15 holds row n0+l15)
        const unsigned wword = mask[(size_t)(n0 + l15) * MASKW + (m0 >> 5)];

#pragma unroll
        for (int r = 0; r < 8; ++r) {
            const unsigned wr = __shfl(wword, r + 8 * half, 32);
            float v0 = ((wr >> l15) & 1u)        ? s[0][r] * scale : -1e30f;
            float v1 = ((wr >> (16 + l15)) & 1u) ? s[1][r] * scale : -1e30f;
            float mx = red_max16(fmaxf(v0, v1));
            float mnew  = fmaxf(mst[r], mx);
            float alpha = __expf(mst[r] - mnew);
            float p0 = (v0 > -1e29f) ? __expf(v0 - mnew) : 0.f;
            float p1 = (v1 > -1e29f) ? __expf(v1 - mnew) : 0.f;
            lst[r] = lst[r] * alpha + red_sum16(p0 + p1);
            mst[r] = mnew;
#pragma unroll
            for (int db = 0; db < 4; ++db) acc[db][r] *= alpha;
            const int row = r + 8 * half;
            pbuf[row * 32 + l15]      = (_Float16)p0;
            pbuf[row * 32 + 16 + l15] = (_Float16)p1;
        }
        __syncthreads();

        // P (16x32) as A-fragment via LDS transpose
        v16h pa;
#pragma unroll
        for (int i = 0; i < 16; ++i)
            pa[i] = pbuf[l15 * 32 + wmma_k16(i, half)];
        __syncthreads();

        // acc += P @ V_tile (32 x 64), packed-B V fragments
#pragma unroll
        for (int db = 0; db < 4; ++db) {
            const v16h bv = *(const v16h*)(vbase + (((size_t)ms * 4 + db) * 32 + lane) * 16);
            acc[db] = __builtin_amdgcn_wmma_f32_16x16x32_f16(
                false, pa, false, bv, (short)0, acc[db], false, false);
        }
    }

    // normalize, transpose through LDS, store packed A-fragments for the Wo GEMM
#pragma unroll
    for (int r = 0; r < 8; ++r) {
        const float inv = 1.f / fmaxf(lst[r], 1e-30f);
        const int row = r + 8 * half;
#pragma unroll
        for (int db = 0; db < 4; ++db)
            pbuf[row * 64 + db * 16 + l15] = (_Float16)(acc[db][r] * inv);
    }
    __syncthreads();

    const int rowtile = blockIdx.z * (NNODES / 16) + blockIdx.x;
#pragma unroll
    for (int ktl = 0; ktl < 2; ++ktl) {
        const int kt = 2 * blockIdx.y + ktl;         // global k-tile (head cols)
        v16h v;
#pragma unroll
        for (int i = 0; i < 16; ++i)
            v[i] = pbuf[l15 * 64 + ktl * 32 + wmma_k16(i, half)];
        *(v16h*)(AttP + (((size_t)rowtile * KT + kt) * 32 + lane) * 16) = v;
    }
}

// ---------------- output projection + residual + LayerNorm ----------------
__global__ void gemm_out_ln_kernel(const _Float16* __restrict__ AttP, const _Float16* __restrict__ WoP,
                                   const float* __restrict__ bo, const float* __restrict__ X,
                                   const float* __restrict__ gamma, const float* __restrict__ beta,
                                   float* __restrict__ out) {
    __shared__ float rs[8][16], rq[8][16];
    const int lane = threadIdx.x & 31;
    const int wave = threadIdx.x >> 5;
    const int half = lane >> 4;
    const int l15  = lane & 15;
    const int row0 = blockIdx.x * 16;
    const int col0 = wave * 64;

    v8f acc[4] = {};
#pragma unroll 1
    for (int kt = 0; kt < KT; ++kt) {
        const v16h a = *(const v16h*)(AttP + (((size_t)blockIdx.x * KT + kt) * 32 + lane) * 16);
#pragma unroll
        for (int db = 0; db < 4; ++db) {
            const int nt = (col0 >> 4) + db;
            const v16h b = *(const v16h*)(WoP + (((size_t)kt * NT + nt) * 32 + lane) * 16);
            acc[db] = __builtin_amdgcn_wmma_f32_16x16x32_f16(
                false, a, false, b, (short)0, acc[db], false, false);
        }
    }

    // bias + residual
#pragma unroll
    for (int db = 0; db < 4; ++db) {
        const int col = col0 + db * 16 + l15;
        const float bv = bo[col];
#pragma unroll
        for (int r = 0; r < 8; ++r)
            acc[db][r] += bv + X[(size_t)(row0 + r + 8 * half) * HIDDEN + col];
    }

    // per-row sums (this wave covers 64 cols; 16 lanes of a half cover them all)
#pragma unroll
    for (int r = 0; r < 8; ++r) {
        float sv = 0.f, sq = 0.f;
#pragma unroll
        for (int db = 0; db < 4; ++db) { float v = acc[db][r]; sv += v; sq += v * v; }
        sv = red_sum16(sv);
        sq = red_sum16(sq);
        if (l15 == 0) { rs[wave][r + 8 * half] = sv; rq[wave][r + 8 * half] = sq; }
    }
    __syncthreads();

#pragma unroll
    for (int r = 0; r < 8; ++r) {
        const int row = r + 8 * half;
        float tot = 0.f, totq = 0.f;
#pragma unroll
        for (int w = 0; w < 8; ++w) { tot += rs[w][row]; totq += rq[w][row]; }
        const float mu   = tot * (1.f / HIDDEN);
        const float var  = totq * (1.f / HIDDEN) - mu * mu;
        const float rinv = rsqrtf(var + 1e-5f);
#pragma unroll
        for (int db = 0; db < 4; ++db) {
            const int col = col0 + db * 16 + l15;
            out[(size_t)(row0 + row) * HIDDEN + col] =
                (acc[db][r] - mu) * rinv * gamma[col] + beta[col];
        }
    }
}

extern "C" void kernel_launch(void* const* d_in, const int* in_sizes, int n_in,
                              void* d_out, int out_size, void* d_ws, size_t ws_size,
                              hipStream_t stream) {
    const float*     x     = (const float*)d_in[0];
    const long long* ei    = (const long long*)d_in[1];
    const float*     Wq    = (const float*)d_in[2];
    const float*     bq    = (const float*)d_in[3];
    const float*     Wk    = (const float*)d_in[4];
    const float*     bk    = (const float*)d_in[5];
    const float*     Wv    = (const float*)d_in[6];
    const float*     bv    = (const float*)d_in[7];
    const float*     Wo    = (const float*)d_in[8];
    const float*     bo    = (const float*)d_in[9];
    const float*     gamma = (const float*)d_in[10];
    const float*     beta  = (const float*)d_in[11];
    const int nedges = in_sizes[1] / 2;

    // ---- workspace layout (all 32B-aligned power-of-two chunks) ----
    const size_t maskBytes = (size_t)NNODES * MASKW * sizeof(unsigned);   // 512 KB
    const size_t bigHalf   = (size_t)ROWS * HIDDEN;                        // 2M halves (4 MB)
    const size_t wHalf     = (size_t)HIDDEN * HIDDEN;                      // 256K halves (512 KB)
    char* p = (char*)d_ws;
    unsigned* mask = (unsigned*)p;            p += maskBytes;
    _Float16* Xp   = (_Float16*)p;            p += bigHalf * 2;
    _Float16* WqP  = (_Float16*)p;            p += wHalf * 2;
    _Float16* WkP  = (_Float16*)p;            p += wHalf * 2;
    _Float16* WvP  = (_Float16*)p;            p += wHalf * 2;
    _Float16* WoP  = (_Float16*)p;            p += wHalf * 2;
    _Float16* Qb   = (_Float16*)p;            p += bigHalf * 2;
    _Float16* Kb   = (_Float16*)p;            p += bigHalf * 2;
    _Float16* Vb   = (_Float16*)p;            p += bigHalf * 2;
    _Float16* KpB  = (_Float16*)p;            p += bigHalf * 2;
    _Float16* VpB  = (_Float16*)p;            p += bigHalf * 2;
    _Float16* AttP = (_Float16*)p;            p += bigHalf * 2;

    hipMemsetAsync(mask, 0, maskBytes, stream);
    build_mask_kernel<<<(nedges + 255) / 256, 256, 0, stream>>>(ei, nedges, mask);

    pack_x_kernel<<<(int)(bigHalf / 256), 256, 0, stream>>>(x, Xp);
    pack_w_kernel<<<(int)(wHalf / 256), 256, 0, stream>>>(Wq, WqP);
    pack_w_kernel<<<(int)(wHalf / 256), 256, 0, stream>>>(Wk, WkP);
    pack_w_kernel<<<(int)(wHalf / 256), 256, 0, stream>>>(Wv, WvP);
    pack_w_kernel<<<(int)(wHalf / 256), 256, 0, stream>>>(Wo, WoP);

    gemm_proj_kernel<<<RT, 256, 0, stream>>>(Xp, WqP, bq, Qb);
    gemm_proj_kernel<<<RT, 256, 0, stream>>>(Xp, WkP, bk, Kb);
    gemm_proj_kernel<<<RT, 256, 0, stream>>>(Xp, WvP, bv, Vb);

    pack_kb_kernel<<<(int)(bigHalf / 256), 256, 0, stream>>>(Kb, KpB);
    pack_vb_kernel<<<(int)(bigHalf / 256), 256, 0, stream>>>(Vb, VpB);

    attention_kernel<<<dim3(NNODES / 16, HEADS, BATCH), 32, 0, stream>>>(Qb, KpB, VpB, mask, AttP);

    gemm_out_ln_kernel<<<RT, 256, 0, stream>>>(AttP, WoP, bo, x, gamma, beta, (float*)d_out);
}
